// FLASH_22660247454340
// MI455X (gfx1250) — matrix operand order repaired and checked by reading, not compiled, not run on hardware
//
#include <hip/hip_runtime.h>
#include <math.h>

typedef __bf16 bf16;
typedef __attribute__((ext_vector_type(16))) __bf16 v16bf;
typedef __attribute__((ext_vector_type(8)))  __bf16 v8bf;
typedef __attribute__((ext_vector_type(8)))  float  v8f;

#define BATCH 4
#define NTOK  4096
#define DIMX  1024
#define HIDX  2048
#define QKDX  128
#define GSZX  256
#define NGRP  16
#define TOKENS (BATCH * NTOK)

// ---------------- WMMA fragment helpers (CDNA5 ISA 7.12.2 layouts) ----------
// A (16x32 bf16): lane L<16 holds row M=L, K={0..7,16..23}; lane L>=16: K={8..15,24..31}
static __device__ __forceinline__ v16bf frag_a(const bf16* base, long stride, int lane) {
  const bf16* p = base + (long)(lane & 15) * stride + ((lane >> 4) << 3);
  v8bf lo = *(const v8bf*)p;
  v8bf hi = *(const v8bf*)(p + 16);
  v16bf f;
#pragma unroll
  for (int i = 0; i < 8; ++i) { f[i] = lo[i]; f[i + 8] = hi[i]; }
  return f;
}
// B (32x16 bf16), stored N-major [n][k]: lane L<16 holds col N=L, K={0..7,16..23}; lane L>=16: K={8..15,24..31}
static __device__ __forceinline__ v16bf frag_b(const bf16* base, long stride, int lane) {
  const bf16* p = base + (long)(lane & 15) * stride + ((lane >> 4) << 3);
  v8bf lo = *(const v8bf*)p;
  v8bf hi = *(const v8bf*)(p + 16);
  v16bf f;
#pragma unroll
  for (int i = 0; i < 8; ++i) { f[i] = lo[i]; f[i + 8] = hi[i]; }
  return f;
}
static __device__ __forceinline__ v8f wmma_bf16(v16bf a, v16bf b, v8f c) {
  return __builtin_amdgcn_wmma_f32_16x16x32_bf16(false, a, false, b, (short)0, c, false, false);
}

// ---------------- CDNA5 async global->LDS copy (ASYNCcnt-tracked) -----------
// LDS address = low 32 bits of generic pointer to __shared__ (ISA 10.2 aperture).
static __device__ __forceinline__ unsigned lds_off(const void* p) {
  return (unsigned)(unsigned long long)p;
}
static __device__ __forceinline__ void async_ld_b128(const bf16* gptr, unsigned ldsoff) {
  asm volatile("global_load_async_to_lds_b128 %0, %1, off"
               :: "v"(ldsoff), "v"((unsigned long long)gptr)
               : "memory");
}
static __device__ __forceinline__ void wait_async0() {
  asm volatile("s_wait_asynccnt 0" ::: "memory");
}

// ---------------- setup kernels ---------------------------------------------
// dst[c][r] = (bf16)src[r][c]  (dst is cols x rows, N-major weight for B operand)
__global__ void transpose_bf16_kernel(const float* __restrict__ src, bf16* __restrict__ dst,
                                      int rows, int cols) {
  long idx = (long)blockIdx.x * 256 + threadIdx.x;
  long total = (long)rows * cols;
  if (idx >= total) return;
  long c = idx / rows, r = idx - c * rows;
  dst[idx] = (bf16)src[r * (long)cols + c];
}

__global__ __launch_bounds__(256) void ln_kernel(const float* __restrict__ x,
                                                 const float* __restrict__ gw,
                                                 const float* __restrict__ bw,
                                                 bf16* __restrict__ out) {
  __shared__ float s1[256], s2[256];
  const int row = blockIdx.x, tid = threadIdx.x;
  const float* xr = x + (long)row * DIMX;
  float4 v = *(const float4*)&xr[tid * 4];
  s1[tid] = v.x + v.y + v.z + v.w;
  s2[tid] = v.x * v.x + v.y * v.y + v.z * v.z + v.w * v.w;
  __syncthreads();
  for (int off = 128; off; off >>= 1) {
    if (tid < off) { s1[tid] += s1[tid + off]; s2[tid] += s2[tid + off]; }
    __syncthreads();
  }
  const float mu  = s1[0] * (1.f / DIMX);
  const float var = s2[0] * (1.f / DIMX) - mu * mu;
  const float inv = rsqrtf(var + 1e-5f);
  bf16* o = out + (long)row * DIMX + tid * 4;
  const float* gp = gw + tid * 4;
  const float* bp = bw + tid * 4;
  o[0] = (bf16)((v.x - mu) * inv * gp[0] + bp[0]);
  o[1] = (bf16)((v.y - mu) * inv * gp[1] + bp[1]);
  o[2] = (bf16)((v.z - mu) * inv * gp[2] + bp[2]);
  o[3] = (bf16)((v.w - mu) * inv * gp[3] + bp[3]);
}

// T5-style relative position bias table, 256x256 fp32
__global__ void bias_kernel(const float* __restrict__ rel, float* __restrict__ bias) {
  const int i = blockIdx.x, j = threadIdx.x;
  const int nrel = i - j;
  const int ret = (nrel < 0) ? 16 : 0;
  const int a = (nrel < 0) ? -nrel : nrel;
  int bucket;
  if (a < 8) {
    bucket = ret + a;
  } else {
    int vl = 8 + (int)(logf((float)a * (1.0f / 8.0f)) * (1.0f / logf(16.0f)) * 8.0f);
    if (vl > 15) vl = 15;
    bucket = ret + vl;
  }
  bias[i * GSZX + j] = rel[bucket] * 11.313708498984761f; // sqrt(QKD)
}

// ---------------- generic 128x128x64 WMMA block-GEMM ------------------------
// Double-buffered: global->LDS tiles move via GLOBAL_LOAD_ASYNC_TO_LDS_B128,
// next K-tile (BK=64) prefetched while 16 WMMAs consume the current one.
struct GemmP {
  const bf16* A; long lda; long aBatch;   // A: M x K, row-major
  const bf16* B; long ldb; long bBatch;   // B: N x K, N-major (pre-transposed)
  int K;
};

template <typename Epi>
__global__ __launch_bounds__(256) void gemm_bf16_kernel(GemmP p, Epi epi) {
  __shared__ bf16 As[2][128 * 64];        // 2 x 16 KB
  __shared__ bf16 Bs[2][128 * 64];        // 2 x 16 KB
  const int tid = threadIdx.x, lane = tid & 31, wave = tid >> 5;
  const int bm = blockIdx.y * 128, bn = blockIdx.x * 128, bat = blockIdx.z;
  const bf16* A = p.A + (long)bat * p.aBatch;
  const bf16* B = p.B + (long)bat * p.bBatch;
  const int wm = (wave >> 1) * 32;   // 4 waves along M (32 rows each)
  const int wn = (wave & 1) * 64;    // 2 waves along N (64 cols each)
  v8f acc[2][4];
#pragma unroll
  for (int i = 0; i < 2; ++i)
#pragma unroll
    for (int j = 0; j < 4; ++j)
#pragma unroll
      for (int r = 0; r < 8; ++r) acc[i][j][r] = 0.f;

  const int lr = tid >> 1;           // 128 rows, 2 threads per row
  const int lc = (tid & 1) * 32;     // each thread covers 32 of 64 k-columns
  const unsigned la[2] = { lds_off(&As[0][lr * 64 + lc]), lds_off(&As[1][lr * 64 + lc]) };
  const unsigned lb[2] = { lds_off(&Bs[0][lr * 64 + lc]), lds_off(&Bs[1][lr * 64 + lc]) };
  const bf16* ga = &A[(long)(bm + lr) * p.lda + lc];
  const bf16* gb = &B[(long)(bn + lr) * p.ldb + lc];

  auto issue = [&](int bufi, int k0) {
#pragma unroll
    for (int c = 0; c < 32; c += 8) {
      async_ld_b128(ga + k0 + c, la[bufi] + 2 * c);
      async_ld_b128(gb + k0 + c, lb[bufi] + 2 * c);
    }
  };

  issue(0, 0);                       // prefetch K-tile 0 into buffer 0
  int buf = 0;
  for (int k0 = 0; k0 < p.K; k0 += 64) {
    wait_async0();                   // own async copies for current buffer landed
    __syncthreads();                 // ... and everyone else's too
    if (k0 + 64 < p.K) issue(buf ^ 1, k0 + 64);  // prefetch next tile
#pragma unroll
    for (int ks = 0; ks < 2; ++ks) { // two 16x16x32 sub-steps per tile
      v16bf af[2], bfr[4];
#pragma unroll
      for (int i = 0; i < 2; ++i)
        af[i] = frag_a(&As[buf][(wm + i * 16) * 64 + ks * 32], 64, lane);
#pragma unroll
      for (int j = 0; j < 4; ++j)
        bfr[j] = frag_b(&Bs[buf][(wn + j * 16) * 64 + ks * 32], 64, lane);
#pragma unroll
      for (int i = 0; i < 2; ++i)
#pragma unroll
        for (int j = 0; j < 4; ++j) acc[i][j] = wmma_bf16(af[i], bfr[j], acc[i][j]);
    }
    buf ^= 1;
  }
  // C/D layout: VGPR r, lane L -> (M = r + 8*(L>>4), N = L&15)
  const int nn = lane & 15, mo = (lane >> 4) * 8;
#pragma unroll
  for (int i = 0; i < 2; ++i)
#pragma unroll
    for (int j = 0; j < 4; ++j)
#pragma unroll
      for (int r = 0; r < 8; ++r)
        epi(bat, bm + wm + i * 16 + mo + r, bn + wn + j * 16 + nn, acc[i][j][r]);
}

// ---------------- epilogues -------------------------------------------------
struct EpiH {            // hv = silu(. + b_h); split v (transposed) / gate
  const float* bh; bf16* vT; bf16* gate;
  __device__ void operator()(int, int m, int n, float v) const {
    float s = v + bh[n];
    s = s / (1.f + __expf(-s));
    if (n < HIDX) {
      const int b = m >> 12, pos = m & (NTOK - 1);
      vT[((long)b * HIDX + n) * NTOK + pos] = (bf16)s;
    } else {
      gate[(long)m * HIDX + (n - HIDX)] = (bf16)s;
    }
  }
};
struct EpiQK {           // qk = silu(. + b_qk); 4 heads via gamma/beta
  const float* bqk; const float* gamma; const float* beta;
  bf16* qq; bf16* lq; bf16* qk2; bf16* lkT;
  __device__ void operator()(int, int m, int n, float v) const {
    float s = v + bqk[n];
    s = s / (1.f + __expf(-s));
    qq[(long)m * QKDX + n]  = (bf16)(s * gamma[n]            + beta[n]);
    lq[(long)m * QKDX + n]  = (bf16)(s * gamma[QKDX + n]     + beta[QKDX + n]);
    qk2[(long)m * QKDX + n] = (bf16)(s * gamma[2 * QKDX + n] + beta[2 * QKDX + n]);
    const int b = m >> 12, pos = m & (NTOK - 1);
    lkT[((long)b * QKDX + n) * NTOK + pos] = (bf16)(s * gamma[3 * QKDX + n] + beta[3 * QKDX + n]);
  }
};
struct EpiKV {           // lin_kv / n, stored transposed [b][e][d]
  bf16* lkvT;
  __device__ void operator()(int bat, int m, int n, float v) const {
    lkvT[((long)bat * HIDX + n) * QKDX + m] = (bf16)(v * (1.0f / NTOK));
  }
};
struct EpiO {            // out = . + b_o + x (fp32 final output)
  const float* bo; const float* x; float* out;
  __device__ void operator()(int, int m, int n, float v) const {
    out[(long)m * DIMX + n] = v + bo[n] + x[(long)m * DIMX + n];
  }
};

// ---------------- fused per-group attention ---------------------------------
// sim = qq qk^T / 256 + bias; attn = relu(sim)^2 (bf16 in 128KB LDS);
// y = gate * (attn @ vg + lin_q @ lin_kv)  -- fused as one K=384 WMMA loop.
__global__ __launch_bounds__(256) void attn_kernel(const bf16* __restrict__ qq,
                                                   const bf16* __restrict__ qkk,
                                                   const bf16* __restrict__ lq,
                                                   const bf16* __restrict__ vT,
                                                   const bf16* __restrict__ lkvT,
                                                   const bf16* __restrict__ gate,
                                                   const float* __restrict__ bias,
                                                   bf16* __restrict__ y) {
  extern __shared__ bf16 attn_s[];           // 256 x 256 bf16 = 128 KB
  const int tid = threadIdx.x, lane = tid & 31, wave = tid >> 5;
  const int bg = blockIdx.x;
  const int b = bg >> 4, g = bg & 15;
  const long t0 = (long)b * NTOK + (long)g * GSZX;   // token base of this group
  const int nn = lane & 15, mo = (lane >> 4) * 8;

  // ---- phase 1: sim -> relu^2 -> LDS (each wave: 2 tile-rows x 16 tile-cols)
#pragma unroll
  for (int tr = 0; tr < 2; ++tr) {
    const int mi = (wave * 2 + tr) * 16;
    v16bf af[4];
#pragma unroll
    for (int kk = 0; kk < 4; ++kk)
      af[kk] = frag_a(qq + (t0 + mi) * QKDX + kk * 32, QKDX, lane);
    for (int tj = 0; tj < 16; ++tj) {
      v8f acc;
#pragma unroll
      for (int r = 0; r < 8; ++r) acc[r] = 0.f;
#pragma unroll
      for (int kk = 0; kk < 4; ++kk)
        acc = wmma_bf16(af[kk], frag_b(qkk + (t0 + tj * 16) * QKDX + kk * 32, QKDX, lane), acc);
#pragma unroll
      for (int r = 0; r < 8; ++r) {
        const int i = mi + mo + r, j = tj * 16 + nn;
        float s = acc[r] * (1.f / GSZX) + bias[i * GSZX + j];
        s = fmaxf(s, 0.f);
        attn_s[i * GSZX + j] = (bf16)(s * s);
      }
    }
  }
  __syncthreads();

  // ---- phase 2: y = gate * ([attn | lin_q] @ [vg ; lin_kv])
  const int mrow = wave * 32;                      // 8 waves x 32 rows = 256
  for (int nc = 0; nc < 16; ++nc) {                // 16 chunks of 128 over HID
    v8f acc[2][8];
#pragma unroll
    for (int i = 0; i < 2; ++i)
#pragma unroll
      for (int j = 0; j < 8; ++j)
#pragma unroll
        for (int r = 0; r < 8; ++r) acc[i][j][r] = 0.f;

    for (int ks = 0; ks < 8; ++ks) {               // attn part, K = 256
      v16bf a0 = frag_a(&attn_s[mrow * GSZX + ks * 32], GSZX, lane);
      v16bf a1 = frag_a(&attn_s[(mrow + 16) * GSZX + ks * 32], GSZX, lane);
#pragma unroll
      for (int j = 0; j < 8; ++j) {
        const long e = nc * 128 + j * 16;
        v16bf bv = frag_b(vT + ((long)b * HIDX + e) * NTOK + g * GSZX + ks * 32, NTOK, lane);
        acc[0][j] = wmma_bf16(a0, bv, acc[0][j]);
        acc[1][j] = wmma_bf16(a1, bv, acc[1][j]);
      }
    }
    for (int ks = 0; ks < 4; ++ks) {               // linear part, K = 128
      v16bf a0 = frag_a(lq + (t0 + mrow) * QKDX + ks * 32, QKDX, lane);
      v16bf a1 = frag_a(lq + (t0 + mrow + 16) * QKDX + ks * 32, QKDX, lane);
#pragma unroll
      for (int j = 0; j < 8; ++j) {
        const long e = nc * 128 + j * 16;
        v16bf bv = frag_b(lkvT + ((long)b * HIDX + e) * QKDX + ks * 32, QKDX, lane);
        acc[0][j] = wmma_bf16(a0, bv, acc[0][j]);
        acc[1][j] = wmma_bf16(a1, bv, acc[1][j]);
      }
    }
#pragma unroll
    for (int i = 0; i < 2; ++i)
#pragma unroll
      for (int j = 0; j < 8; ++j)
#pragma unroll
        for (int r = 0; r < 8; ++r) {
          const long token = t0 + mrow + i * 16 + mo + r;
          const int  e = nc * 128 + j * 16 + nn;
          const float gv = (float)gate[token * HIDX + e];
          y[token * HIDX + e] = (bf16)(gv * acc[i][j][r]);
        }
  }
}

// ---------------- launcher --------------------------------------------------
extern "C" void kernel_launch(void* const* d_in, const int* in_sizes, int n_in,
                              void* d_out, int out_size, void* d_ws, size_t ws_size,
                              hipStream_t stream) {
  (void)in_sizes; (void)n_in; (void)out_size; (void)ws_size;
  const float* x        = (const float*)d_in[0];
  const float* norm_g   = (const float*)d_in[1];
  const float* norm_b   = (const float*)d_in[2];
  const float* W_h      = (const float*)d_in[3];
  const float* b_h      = (const float*)d_in[4];
  const float* W_qk     = (const float*)d_in[5];
  const float* b_qk     = (const float*)d_in[6];
  const float* qk_gamma = (const float*)d_in[7];
  const float* qk_beta  = (const float*)d_in[8];
  const float* rel_emb  = (const float*)d_in[9];
  const float* W_o      = (const float*)d_in[10];
  const float* b_o      = (const float*)d_in[11];
  float* out = (float*)d_out;

  char* ws = (char*)d_ws;
  size_t off = 0;
  auto take = [&](size_t bytes) -> char* {
    char* p = ws + off;
    off += (bytes + 255) & ~(size_t)255;
    return p;
  };
  bf16* normed = (bf16*)take((size_t)TOKENS * DIMX * 2);           // 33.5 MB
  bf16* W_hT   = (bf16*)take((size_t)2 * HIDX * DIMX * 2);         //  8.4 MB
  bf16* W_qkT  = (bf16*)take((size_t)QKDX * DIMX * 2);             //  0.3 MB
  bf16* W_oT   = (bf16*)take((size_t)DIMX * HIDX * 2);             //  4.2 MB
  bf16* vT     = (bf16*)take((size_t)BATCH * HIDX * NTOK * 2);     // 67.1 MB
  bf16* gate   = (bf16*)take((size_t)TOKENS * HIDX * 2);           // 67.1 MB
  bf16* qq     = (bf16*)take((size_t)TOKENS * QKDX * 2);           //  4.2 MB
  bf16* qk2    = (bf16*)take((size_t)TOKENS * QKDX * 2);           //  4.2 MB
  bf16* lq     = (bf16*)take((size_t)TOKENS * QKDX * 2);           //  4.2 MB
  bf16* lkT    = (bf16*)take((size_t)BATCH * QKDX * NTOK * 2);     //  4.2 MB
  bf16* lkvT   = (bf16*)take((size_t)BATCH * HIDX * QKDX * 2);     //  2.1 MB
  float* bias  = (float*)take((size_t)GSZX * GSZX * 4);            //  0.3 MB
  bf16* y      = (bf16*)take((size_t)TOKENS * HIDX * 2);           // 67.1 MB

  // setup: weight casts/transposes, layernorm, bias table
  transpose_bf16_kernel<<<(DIMX * 2 * HIDX + 255) / 256, 256, 0, stream>>>(W_h, W_hT, DIMX, 2 * HIDX);
  transpose_bf16_kernel<<<(DIMX * QKDX + 255) / 256, 256, 0, stream>>>(W_qk, W_qkT, DIMX, QKDX);
  transpose_bf16_kernel<<<(HIDX * DIMX + 255) / 256, 256, 0, stream>>>(W_o, W_oT, HIDX, DIMX);
  ln_kernel<<<TOKENS, 256, 0, stream>>>(x, norm_g, norm_b, normed);
  bias_kernel<<<GSZX, GSZX, 0, stream>>>(rel_emb, bias);

  // hv = silu(normed @ W_h + b_h) -> vT / gate   (M=16384, N=4096, K=1024)
  GemmP ph{normed, DIMX, 0, W_hT, DIMX, 0, DIMX};
  gemm_bf16_kernel<<<dim3(2 * HIDX / 128, TOKENS / 128, 1), 256, 0, stream>>>(ph, EpiH{b_h, vT, gate});

  // qk = silu(normed @ W_qk + b_qk) -> 4 heads   (M=16384, N=128, K=1024)
  GemmP pqk{normed, DIMX, 0, W_qkT, DIMX, 0, DIMX};
  gemm_bf16_kernel<<<dim3(1, TOKENS / 128, 1), 256, 0, stream>>>(
      pqk, EpiQK{b_qk, qk_gamma, qk_beta, qq, lq, qk2, lkT});

  // lin_kv[b] = lin_k^T @ v / n                  (M=128, N=2048, K=4096, batched)
  GemmP pkv{lkT, NTOK, (long)QKDX * NTOK, vT, NTOK, (long)HIDX * NTOK, NTOK};
  gemm_bf16_kernel<<<dim3(HIDX / 128, 1, BATCH), 256, 0, stream>>>(pkv, EpiKV{lkvT});

  // fused quadratic + linear attention + gate -> y
  attn_kernel<<<BATCH * NGRP, 256, (size_t)GSZX * GSZX * 2, stream>>>(
      qq, qk2, lq, vT, lkvT, gate, bias, y);

  // out = y @ W_o + b_o + x                      (M=16384, N=1024, K=2048)
  GemmP po{y, HIDX, 0, W_oT, HIDX, 0, HIDX};
  gemm_bf16_kernel<<<dim3(DIMX / 128, TOKENS / 128, 1), 256, 0, stream>>>(po, EpiO{b_o, x, out});
}
